// TBStars2MoEDecoderLayer_18614388261195
// MI455X (gfx1250) — matrix operand order repaired and checked by reading, not compiled, not run on hardware
//
#include <hip/hip_runtime.h>
#include <hip/hip_bf16.h>

// ---------------------------------------------------------------------------
// Types for CDNA5 WMMA (wave32, V_WMMA_F32_16X16X32_F16)
// ---------------------------------------------------------------------------
typedef _Float16 h4_t  __attribute__((ext_vector_type(4)));
typedef _Float16 h8_t  __attribute__((ext_vector_type(8)));
typedef _Float16 v16h  __attribute__((ext_vector_type(16)));
typedef float    v8f   __attribute__((ext_vector_type(8)));
typedef float    f4_t  __attribute__((ext_vector_type(4)));

#define TSEQ 2048
#define HDIM 2048
#define NHEAD 16
#define HEADD 128
#define NEXP 16
#define FMOE 1024

__device__ __forceinline__ v8f wmma_f16(v16h a, v16h b, v8f c) {
    return __builtin_amdgcn_wmma_f32_16x16x32_f16(
        false, a, false, b, (short)0, c, false, false);
}

// A fragment from LDS row (row-major halves, stride 40): lane-half kOff in {0,16}
// elements 0..15 = K kOff..kOff+15 (contiguous)
__device__ __forceinline__ v16h frag_a_lds(const _Float16* row, int kOff) {
    h8_t lo = *(const h8_t*)(row + kOff);
    h8_t hi = *(const h8_t*)(row + kOff + 8);
    v16h r;
#pragma unroll
    for (int i = 0; i < 8; ++i) { r[i] = lo[i]; r[i + 8] = hi[i]; }
    return r;
}

// B fragment from LDS row (column of B stored contiguously): 16 contiguous K
// starting at koffB = 16*half
__device__ __forceinline__ v16h frag_b_lds(const _Float16* row, int koffB) {
    h8_t lo = *(const h8_t*)(row + koffB);
    h8_t hi = *(const h8_t*)(row + koffB + 8);
    v16h r;
#pragma unroll
    for (int i = 0; i < 8; ++i) { r[i] = lo[i]; r[i + 8] = hi[i]; }
    return r;
}

// A fragment directly from global fp32 row (with scale folded in)
__device__ __forceinline__ v16h frag_a_g(const float* p, int kOff, float scale) {
    f4_t a = *(const f4_t*)(p + kOff);
    f4_t b = *(const f4_t*)(p + kOff + 4);
    f4_t c = *(const f4_t*)(p + kOff + 8);
    f4_t d = *(const f4_t*)(p + kOff + 12);
    v16h r;
#pragma unroll
    for (int i = 0; i < 4; ++i) {
        r[i]      = (_Float16)(a[i] * scale);
        r[4 + i]  = (_Float16)(b[i] * scale);
        r[8 + i]  = (_Float16)(c[i] * scale);
        r[12 + i] = (_Float16)(d[i] * scale);
    }
    return r;
}

// B fragment directly from global fp32 row (16 contiguous values)
__device__ __forceinline__ v16h frag_b_g(const float* p, int koffB) {
    f4_t a = *(const f4_t*)(p + koffB);
    f4_t b = *(const f4_t*)(p + koffB + 4);
    f4_t c = *(const f4_t*)(p + koffB + 8);
    f4_t d = *(const f4_t*)(p + koffB + 12);
    v16h r;
#pragma unroll
    for (int i = 0; i < 4; ++i) {
        r[i]      = (_Float16)a[i];
        r[4 + i]  = (_Float16)b[i];
        r[8 + i]  = (_Float16)c[i];
        r[12 + i] = (_Float16)d[i];
    }
    return r;
}

__device__ __forceinline__ h8_t pack8(f4_t lo, f4_t hi) {
    h8_t r;
#pragma unroll
    for (int i = 0; i < 4; ++i) { r[i] = (_Float16)lo[i]; r[4 + i] = (_Float16)hi[i]; }
    return r;
}

// ---------------------------------------------------------------------------
// Generic tiled WMMA GEMM:  C[M,N] = A[M,K] * B[K,N]  (+ add1 + add2)
// 128x128 tile per block (256 threads = 8 waves, each wave 64x32).
// Optional: per-expert row windows (offs), A row gather, C row scatter.
// fp32 global operands converted to f16 in LDS staging.
// ---------------------------------------------------------------------------
__global__ __launch_bounds__(256) void gemm_f16_wmma(
    const float* __restrict__ A, int lda,
    const float* __restrict__ B, int ldb, long long bstride,
    float* __restrict__ C, int ldc,
    int Mfixed, int Kdim,
    const int* __restrict__ expert_offs,
    const int* __restrict__ a_gather,
    const int* __restrict__ c_scatter,
    const float* __restrict__ add1,
    const float* __restrict__ add2)
{
    __shared__ _Float16 As[128][40];
    __shared__ _Float16 Bs[128][40];   // stores B transposed: Bs[n][k]

    int z = blockIdx.z;
    int rowstart = 0, M = Mfixed;
    if (expert_offs) { rowstart = expert_offs[z]; M = expert_offs[z + 1] - rowstart; }
    int bm = blockIdx.y * 128;
    if (bm >= M) return;
    int bn = blockIdx.x * 128;
    const float* Bz = B + (long long)z * bstride;

    int tid = threadIdx.x;
    int wave = tid >> 5, lane = tid & 31;
    int waveM = wave >> 2, waveN = wave & 3;      // 2 x 4 wave grid
    int lh = lane >> 4, l15 = lane & 15;
    int kOffA = lh * 16, kOffB = lh * 16;

    v8f zacc = {};
    v8f acc[4][2];
#pragma unroll
    for (int i = 0; i < 4; ++i)
#pragma unroll
        for (int j = 0; j < 2; ++j) acc[i][j] = zacc;

    // staging roles (fixed per thread)
    int r_a = tid >> 1;                // 0..127 A row in tile
    int c_a = (tid & 1) << 4;          // 0 or 16
    int arow = -1;
    if (bm + r_a < M)
        arow = a_gather ? a_gather[rowstart + bm + r_a] : (rowstart + bm + r_a);
    int kg  = tid >> 5;                // 0..7   B k-group (4 rows)
    int cb4 = (lane) << 2;             // 0..124 B column group (4 cols)

    for (int k0 = 0; k0 < Kdim; k0 += 32) {
        // stage A (convert fp32 -> f16), packed 16B LDS stores
        if (arow >= 0) {
            const float* src = A + (long long)arow * lda + k0 + c_a;
            f4_t v0 = *(const f4_t*)(src);
            f4_t v1 = *(const f4_t*)(src + 4);
            f4_t v2 = *(const f4_t*)(src + 8);
            f4_t v3 = *(const f4_t*)(src + 12);
            *(h8_t*)&As[r_a][c_a]     = pack8(v0, v1);
            *(h8_t*)&As[r_a][c_a + 8] = pack8(v2, v3);
            __builtin_prefetch(src + 32, 0, 3);          // next k-tile
        } else if (k0 == 0) {                            // zero rows persist
            h8_t zz = {};
            *(h8_t*)&As[r_a][c_a]     = zz;
            *(h8_t*)&As[r_a][c_a + 8] = zz;
        }
        // stage B transposed: 4x4 micro-tile per thread, b64 LDS stores
        {
            const float* src = Bz + (long long)(k0 + 4 * kg) * ldb + bn + cb4;
            f4_t r0 = *(const f4_t*)(src);
            f4_t r1 = *(const f4_t*)(src + ldb);
            f4_t r2 = *(const f4_t*)(src + 2 * ldb);
            f4_t r3 = *(const f4_t*)(src + 3 * ldb);
            __builtin_prefetch(src + (long long)32 * ldb, 0, 3);
#pragma unroll
            for (int j = 0; j < 4; ++j) {
                h4_t h;
                h[0] = (_Float16)r0[j]; h[1] = (_Float16)r1[j];
                h[2] = (_Float16)r2[j]; h[3] = (_Float16)r3[j];
                *(h4_t*)&Bs[cb4 + j][4 * kg] = h;
            }
        }
        __syncthreads();

        v16h af[4], bf[2];
#pragma unroll
        for (int tm = 0; tm < 4; ++tm)
            af[tm] = frag_a_lds(&As[waveM * 64 + tm * 16 + l15][0], kOffA);
#pragma unroll
        for (int tn = 0; tn < 2; ++tn)
            bf[tn] = frag_b_lds(&Bs[waveN * 32 + tn * 16 + l15][0], kOffB);
#pragma unroll
        for (int tm = 0; tm < 4; ++tm)
#pragma unroll
            for (int tn = 0; tn < 2; ++tn)
                acc[tm][tn] = wmma_f16(af[tm], bf[tn], acc[tm][tn]);
        __syncthreads();
    }

    // epilogue: C layout — VGPR r = row (r + 8*half), lane&15 = col
#pragma unroll
    for (int tm = 0; tm < 4; ++tm) {
#pragma unroll
        for (int r = 0; r < 8; ++r) {
            int m = bm + waveM * 64 + tm * 16 + 8 * lh + r;
            if (m >= M) continue;
            int crow = c_scatter ? c_scatter[rowstart + m] : (rowstart + m);
#pragma unroll
            for (int tn = 0; tn < 2; ++tn) {
                int col = bn + waveN * 32 + tn * 16 + l15;
                long long o = (long long)crow * ldc + col;
                float v = acc[tm][tn][r];
                if (add1) v += add1[o];
                if (add2) v += add2[o];
                C[o] = v;
            }
        }
    }
}

// ---------------------------------------------------------------------------
// Flash-style causal attention, one block = 128 query rows of one head.
// ---------------------------------------------------------------------------
__global__ __launch_bounds__(256) void attn_wmma(
    const float* __restrict__ qkv, float* __restrict__ ctx)
{
    __shared__ _Float16 Vt[128][40];       // V block transposed: Vt[d][key]
    __shared__ _Float16 Pl[8][16][40];     // per-wave P tile (C->A relayout)

    int head = blockIdx.x;
    int qstart = blockIdx.y * 128;
    int tid = threadIdx.x;
    int wave = tid >> 5, lane = tid & 31;
    int lh = lane >> 4, l15 = lane & 15;
    int kOffA = lh * 16, kOffB = lh * 16;
    const float scale = 0.08838834764831845f;   // 1/sqrt(128)

    // Q fragments (rows qstart + wave*16 .. +15), scale folded in
    int qrow = qstart + wave * 16 + l15;
    const float* qptr = qkv + (long long)qrow * 6144 + head * HEADD;
    v16h qf[4];
#pragma unroll
    for (int kc = 0; kc < 4; ++kc)
        qf[kc] = frag_a_g(qptr + kc * 32, kOffA, scale);

    v8f zacc = {};
    float m_i[8], l_i[8];
    v8f o[8];
#pragma unroll
    for (int r = 0; r < 8; ++r) { m_i[r] = -1e30f; l_i[r] = 0.0f; }
#pragma unroll
    for (int tn = 0; tn < 8; ++tn) o[tn] = zacc;

    int kg = tid >> 5;         // V staging: key group (4 keys)
    int d4 = (lane) << 2;      // 4 dims

    int kend = qstart + 128;
    for (int kb0 = 0; kb0 < kend; kb0 += 32) {
        __syncthreads();
        // cooperative V staging: 4x4 micro-tile transpose, b64 LDS stores
        {
            const float* vsrc = qkv + (long long)(kb0 + 4 * kg) * 6144 + 4096 + head * HEADD + d4;
            f4_t r0 = *(const f4_t*)(vsrc);
            f4_t r1 = *(const f4_t*)(vsrc + 6144);
            f4_t r2 = *(const f4_t*)(vsrc + 2 * 6144);
            f4_t r3 = *(const f4_t*)(vsrc + 3 * 6144);
#pragma unroll
            for (int j = 0; j < 4; ++j) {
                h4_t h;
                h[0] = (_Float16)r0[j]; h[1] = (_Float16)r1[j];
                h[2] = (_Float16)r2[j]; h[3] = (_Float16)r3[j];
                *(h4_t*)&Vt[d4 + j][4 * kg] = h;
            }
        }
        __syncthreads();

        // S = Q * K^T : B fragments directly from global K rows
        v8f s0 = zacc, s1 = zacc;
        int key0 = kb0 + l15, key1 = kb0 + 16 + l15;
        const float* kp0 = qkv + (long long)key0 * 6144 + 2048 + head * HEADD;
        const float* kp1 = qkv + (long long)key1 * 6144 + 2048 + head * HEADD;
#pragma unroll
        for (int kc = 0; kc < 4; ++kc) {
            v16h b0 = frag_b_g(kp0 + kc * 32, kOffB);
            v16h b1 = frag_b_g(kp1 + kc * 32, kOffB);
            s0 = wmma_f16(qf[kc], b0, s0);
            s1 = wmma_f16(qf[kc], b1, s1);
        }

        // streaming softmax update (rows rowbase..rowbase+7 for this half)
        int rowbase = qstart + wave * 16 + 8 * lh;
#pragma unroll
        for (int r = 0; r < 8; ++r) {
            int row = rowbase + r;
            float a0 = (key0 <= row) ? s0[r] : -1e30f;
            float a1 = (key1 <= row) ? s1[r] : -1e30f;
            float mx = fmaxf(a0, a1);
#pragma unroll
            for (int d = 1; d < 16; d <<= 1) mx = fmaxf(mx, __shfl_xor(mx, d, 32));
            float mn = fmaxf(m_i[r], mx);
            float p0 = __expf(a0 - mn);
            float p1 = __expf(a1 - mn);
            float ps = p0 + p1;
#pragma unroll
            for (int d = 1; d < 16; d <<= 1) ps += __shfl_xor(ps, d, 32);
            float sc = __expf(m_i[r] - mn);
            l_i[r] = l_i[r] * sc + ps;
            m_i[r] = mn;
#pragma unroll
            for (int tn = 0; tn < 8; ++tn) o[tn][r] *= sc;
            Pl[wave][8 * lh + r][l15]      = (_Float16)p0;
            Pl[wave][8 * lh + r][16 + l15] = (_Float16)p1;
        }
        // wave-local: make P stores visible to this wave's loads
        asm volatile("s_wait_dscnt 0x0" ::: "memory");

        v16h pA = frag_a_lds(&Pl[wave][l15][0], kOffA);
#pragma unroll
        for (int tn = 0; tn < 8; ++tn) {
            v16h bV = frag_b_lds(&Vt[tn * 16 + l15][0], kOffB);
            o[tn] = wmma_f16(pA, bV, o[tn]);
        }
    }

    // write ctx (T x H), col = head*128 + d
    float* cbase = ctx + (long long)(qstart + wave * 16) * HDIM + head * HEADD;
#pragma unroll
    for (int tn = 0; tn < 8; ++tn)
#pragma unroll
        for (int r = 0; r < 8; ++r) {
            int row = 8 * lh + r;
            cbase[(long long)row * HDIM + tn * 16 + l15] = o[tn][r] / l_i[r];
        }
}

// ---------------------------------------------------------------------------
// Small helper kernels
// ---------------------------------------------------------------------------
__global__ __launch_bounds__(256) void rmsnorm_k(
    const float* __restrict__ x, const float* __restrict__ g, float* __restrict__ o)
{
    __shared__ float red[256];
    int t = blockIdx.x;
    const float* xp = x + (long long)t * HDIM;
    float ss = 0.0f;
    for (int i = threadIdx.x; i < HDIM; i += 256) { float v = xp[i]; ss += v * v; }
    red[threadIdx.x] = ss; __syncthreads();
    for (int s = 128; s > 0; s >>= 1) {
        if (threadIdx.x < s) red[threadIdx.x] += red[threadIdx.x + s];
        __syncthreads();
    }
    float rs = rsqrtf(red[0] / (float)HDIM + 1e-6f);
    float* op = o + (long long)t * HDIM;
    for (int i = threadIdx.x; i < HDIM; i += 256) op[i] = xp[i] * rs * g[i];
}

__global__ __launch_bounds__(256) void rope_k(float* __restrict__ qkv)
{
    int idx = blockIdx.x * 256 + threadIdx.x;       // T*NH*64
    int t = idx >> 10; int rem = idx & 1023; int h = rem >> 6; int j = rem & 63;
    float ang = (float)t * __expf(-0.14391156f * (float)j);  // t * 10000^(-j/64)
    float c = cosf(ang), s = sinf(ang);
    long long base = (long long)t * 6144 + h * HEADD + j;
    float q1 = qkv[base], q2 = qkv[base + 64];
    qkv[base]      = q1 * c - q2 * s;
    qkv[base + 64] = q1 * s + q2 * c;
    long long kb = base + 2048;
    float k1 = qkv[kb], k2 = qkv[kb + 64];
    qkv[kb]      = k1 * c - k2 * s;
    qkv[kb + 64] = k1 * s + k2 * c;
}

__global__ __launch_bounds__(256) void router_k(
    const float* __restrict__ h2, const float* __restrict__ gw,
    float* __restrict__ topw, int* __restrict__ topi, int* __restrict__ counts)
{
    __shared__ float buf[256];
    __shared__ float lg[NEXP];
    int t = blockIdx.x;
    float p[NEXP];
#pragma unroll
    for (int e = 0; e < NEXP; ++e) p[e] = 0.0f;
    for (int i = threadIdx.x; i < HDIM; i += 256) {
        float xv = h2[(long long)t * HDIM + i];
        const float* grow = gw + (long long)i * NEXP;
#pragma unroll
        for (int e = 0; e < NEXP; ++e) p[e] += xv * grow[e];
    }
    for (int e = 0; e < NEXP; ++e) {
        buf[threadIdx.x] = p[e]; __syncthreads();
        for (int s = 128; s > 0; s >>= 1) {
            if (threadIdx.x < s) buf[threadIdx.x] += buf[threadIdx.x + s];
            __syncthreads();
        }
        if (threadIdx.x == 0) lg[e] = buf[0];
        __syncthreads();
    }
    if (threadIdx.x == 0) {
        int i1 = 0;
        for (int e = 1; e < NEXP; ++e) if (lg[e] > lg[i1]) i1 = e;
        int i2 = -1;
        for (int e = 0; e < NEXP; ++e)
            if (e != i1 && (i2 < 0 || lg[e] > lg[i2])) i2 = e;
        float d = __expf(lg[i2] - lg[i1]);
        float w1 = 1.0f / (1.0f + d);
        topi[2 * t] = i1; topi[2 * t + 1] = i2;
        topw[2 * t] = w1; topw[2 * t + 1] = d * w1;
        atomicAdd(&counts[i1], 1);
        atomicAdd(&counts[i2], 1);
    }
}

__global__ void zero_counts_k(int* counts, int* fill)
{
    int i = threadIdx.x;
    if (i < NEXP) { counts[i] = 0; fill[i] = 0; }
}

__global__ void scan_k(const int* __restrict__ counts, int* __restrict__ offs)
{
    if (threadIdx.x == 0) {
        int a = 0;
        for (int e = 0; e < NEXP; ++e) { offs[e] = a; a += counts[e]; }
        offs[NEXP] = a;
    }
}

__global__ __launch_bounds__(256) void fill_k(
    const int* __restrict__ topi, const int* __restrict__ offs,
    int* __restrict__ fill, int* __restrict__ etok, int* __restrict__ eout)
{
    int t = blockIdx.x * 256 + threadIdx.x;
    if (t < TSEQ) {
#pragma unroll
        for (int s = 0; s < 2; ++s) {
            int e = topi[2 * t + s];
            int pos = atomicAdd(&fill[e], 1);
            int id = offs[e] + pos;
            etok[id] = t;
            eout[id] = 2 * t + s;
        }
    }
}

__global__ __launch_bounds__(256) void silu_mul_k(
    const float* __restrict__ in, float* __restrict__ out, int F2)
{
    int f = blockIdx.x * 256 + threadIdx.x;
    long long r = blockIdx.y;
    float g = in[r * 2 * F2 + f];
    float u = in[r * 2 * F2 + F2 + f];
    out[r * F2 + f] = (g / (1.0f + __expf(-g))) * u;
}

__global__ __launch_bounds__(256) void combine_k(
    const float* __restrict__ teh, const float* __restrict__ topw, float* __restrict__ mo)
{
    int hcol = blockIdx.x * 256 + threadIdx.x;
    long long t = blockIdx.y;
    mo[t * HDIM + hcol] = topw[2 * t]     * teh[(2 * t)     * HDIM + hcol]
                        + topw[2 * t + 1] * teh[(2 * t + 1) * HDIM + hcol];
}

// ---------------------------------------------------------------------------
// Launcher
// ---------------------------------------------------------------------------
extern "C" void kernel_launch(void* const* d_in, const int* in_sizes, int n_in,
                              void* d_out, int out_size, void* d_ws, size_t ws_size,
                              hipStream_t stream)
{
    (void)in_sizes; (void)n_in; (void)out_size; (void)ws_size;
    const float* x       = (const float*)d_in[0];
    const float* gamma1  = (const float*)d_in[1];
    const float* gamma2  = (const float*)d_in[2];
    const float* w_qkv   = (const float*)d_in[3];
    const float* w_o     = (const float*)d_in[4];
    const float* gate_w  = (const float*)d_in[5];
    const float* moe_up  = (const float*)d_in[6];
    const float* moe_dn  = (const float*)d_in[7];
    const float* w13     = (const float*)d_in[8];
    const float* wdown   = (const float*)d_in[9];
    float* out = (float*)d_out;

    float* ws = (float*)d_ws;
    size_t off = 0;
    auto take = [&](size_t n) { float* p = ws + off; off += (n + 255) & ~(size_t)255; return p; };

    const size_t TH = (size_t)TSEQ * HDIM;           // 4,194,304
    float* regA  = take(16777216);                   // qkv+ctx / gu+act / sg
    float* qkv   = regA;                             // T*3H
    float* ctx   = regA + (size_t)TSEQ * 3 * HDIM;   // T*H
    float* gu    = regA;                             // 2T * 2F
    float* act   = regA + (size_t)2 * TSEQ * 2 * FMOE; // 2T * F
    float* sg    = regA;                             // T * 8192
    float* hbuf  = take(TH);                         // h, later moe_out
    float* moeo  = hbuf;
    float* x1    = take(TH);
    float* h2    = take(TH);
    float* regE  = take((size_t)2 * TSEQ * HDIM);    // teh, later s_act
    float* teh   = regE;
    float* s_act = regE;
    float* topw  = take(2 * TSEQ);
    int* ibase = (int*)(ws + off);
    int* topi  = ibase;
    int* etok  = topi + 2 * TSEQ;
    int* eout  = etok + 2 * TSEQ;
    int* counts = eout + 2 * TSEQ;
    int* fill   = counts + NEXP;
    int* offs   = fill + NEXP;

    // 1. h = rmsnorm(x, gamma1)
    rmsnorm_k<<<TSEQ, 256, 0, stream>>>(x, gamma1, hbuf);

    // 2. qkv = h @ w_qkv   (2048 x 6144 x 2048)
    gemm_f16_wmma<<<dim3(48, 16, 1), 256, 0, stream>>>(
        hbuf, HDIM, w_qkv, 3 * HDIM, 0, qkv, 3 * HDIM,
        TSEQ, HDIM, nullptr, nullptr, nullptr, nullptr, nullptr);

    // 3. rope in-place on q,k
    rope_k<<<(TSEQ * NHEAD * 64) / 256, 256, 0, stream>>>(qkv);

    // 4. causal attention -> ctx
    attn_wmma<<<dim3(NHEAD, TSEQ / 128, 1), 256, 0, stream>>>(qkv, ctx);

    // 5. x1 = x + ctx @ w_o
    gemm_f16_wmma<<<dim3(16, 16, 1), 256, 0, stream>>>(
        ctx, HDIM, w_o, HDIM, 0, x1, HDIM,
        TSEQ, HDIM, nullptr, nullptr, nullptr, x, nullptr);

    // 6. h2 = rmsnorm(x1, gamma2)
    rmsnorm_k<<<TSEQ, 256, 0, stream>>>(x1, gamma2, h2);

    // 7. routing
    zero_counts_k<<<1, 64, 0, stream>>>(counts, fill);
    router_k<<<TSEQ, 256, 0, stream>>>(h2, gate_w, topw, topi, counts);
    scan_k<<<1, 32, 0, stream>>>(counts, offs);
    fill_k<<<TSEQ / 256, 256, 0, stream>>>(topi, offs, fill, etok, eout);

    // 8. routed up-proj: gu[entry] = h2[token] @ moe_up[e]  (per expert)
    gemm_f16_wmma<<<dim3(16, 16, NEXP), 256, 0, stream>>>(
        h2, HDIM, moe_up, 2 * FMOE, (long long)HDIM * 2 * FMOE, gu, 2 * FMOE,
        0, HDIM, offs, etok, nullptr, nullptr, nullptr);

    // 9. act = silu(g) * u   (4096 rows x 1024)
    silu_mul_k<<<dim3(FMOE / 256, 2 * TSEQ, 1), 256, 0, stream>>>(gu, act, FMOE);

    // 10. routed down-proj, scatter to (token,slot) rows of teh
    gemm_f16_wmma<<<dim3(16, 16, NEXP), 256, 0, stream>>>(
        act, FMOE, moe_dn, HDIM, (long long)FMOE * HDIM, teh, HDIM,
        0, FMOE, offs, nullptr, eout, nullptr, nullptr);

    // 11. moe_out = w0*teh[2t] + w1*teh[2t+1]
    combine_k<<<dim3(HDIM / 256, TSEQ, 1), 256, 0, stream>>>(teh, topw, moeo);

    // 12. sg = h2 @ shared_w13   (2048 x 8192 x 2048)
    gemm_f16_wmma<<<dim3(64, 16, 1), 256, 0, stream>>>(
        h2, HDIM, w13, 8192, 0, sg, 8192,
        TSEQ, HDIM, nullptr, nullptr, nullptr, nullptr, nullptr);

    // 13. s_act = silu(sgate) * sup   (2048 x 4096)
    silu_mul_k<<<dim3(4096 / 256, TSEQ, 1), 256, 0, stream>>>(sg, s_act, 4096);

    // 14. out = x1 + moe_out + s_act @ shared_down   (2048 x 2048 x 4096)
    gemm_f16_wmma<<<dim3(16, 16, 1), 256, 0, stream>>>(
        s_act, 4096, wdown, HDIM, 0, out, HDIM,
        TSEQ, 4096, nullptr, nullptr, nullptr, x1, moeo);
}
